// User_Caption_5334349381922
// MI455X (gfx1250) — matrix operand order repaired and checked by reading, not compiled, not run on hardware
//
#include <hip/hip_runtime.h>
#include <math.h>

typedef __attribute__((ext_vector_type(16))) _Float16 v16h;
typedef __attribute__((ext_vector_type(8)))  _Float16 v8h;
typedef __attribute__((ext_vector_type(8)))  float    v8f;

#define H_DIM   1024
#define W_DIM   512
#define IMG_DIM 2048
#define U_DIM   256
#define VOC     32000
#define CAP_LEN 64
#define GATES   4096
#define SOS     (VOC - 2)
#define XCH     (W_DIM >> 5)   // 16 k-chunks over x
#define HCH     (H_DIM >> 5)   // 32 k-chunks over h
#define NTILES_P (VOC / 16)    // 2000 logit tiles

// -------------------------------------------------------------------------
// Pack f32 weight matrix [rows x K] into per-wave WMMA A-fragment layout
// (16x32 f16 tiles). One wave per (tile, k-chunk); each lane's 16 halves are
// stored contiguously so the consumer's loads are fully coalesced.
// Fragment layout (16-bit A 16x32, contiguous k order):
//   lanes 0-15 : row = tile*16+lane,    halves 0-15 -> K 0-15
//   lanes 16-31: row = tile*16+lane-16, halves 0-15 -> K 16-31
// -------------------------------------------------------------------------
__global__ void pack_w_kernel(const float* __restrict__ W, _Float16* __restrict__ P,
                              int rows, int K) {
  const int chunks = K >> 5;
  const int tiles  = rows >> 4;
  const int wave = (blockIdx.x * blockDim.x + threadIdx.x) >> 5;
  const int lane = threadIdx.x & 31;
  if (wave >= tiles * chunks) return;
  const int tile  = wave / chunks;
  const int chunk = wave - tile * chunks;
  const int m  = lane & 15;
  const int hi = lane >> 4;
  const float* src = W + (size_t)(tile * 16 + m) * K + chunk * 32;
  _Float16 frag[16];
#pragma unroll
  for (int h = 0; h < 16; ++h) {
    int k = hi * 16 + h;
    frag[h] = (_Float16)src[k];
  }
  _Float16* dst = P + ((size_t)wave * 32 + lane) * 16;
  *(v8h*)dst       = *(v8h*)frag;
  *(v8h*)(dst + 8) = *(v8h*)(frag + 8);
}

// ------------------------------------------------------------------
// Prologue: bias = b_ih + b_hh; zero h/c; wid = SOS; ticket = 0.
// grid 16 x 256 = 4096 threads.
// ------------------------------------------------------------------
__global__ void prep_kernel(const float* __restrict__ b_ih,
                            const float* __restrict__ b_hh,
                            float* __restrict__ bias,
                            _Float16* __restrict__ h16, float* __restrict__ c,
                            int* __restrict__ wid, int* __restrict__ ticket) {
  const int j = blockIdx.x * blockDim.x + threadIdx.x;
  bias[j] = b_ih[j] + b_hh[j];
  if (j < H_DIM) { h16[j] = (_Float16)0.0f; c[j] = 0.0f; }
  if (j == 0) { *wid = SOS; *ticket = 0; }
}

// ------------------------------------------------------------------
// x0 = WI_w @ concat(img, wei_user[uid]) + WI_b  -> f16
// One wave per output row; input vector staged in LDS.
// ------------------------------------------------------------------
__global__ void x0_kernel(const float* __restrict__ img,
                          const float* __restrict__ wei_user,
                          const int* __restrict__ uid,
                          const float* __restrict__ WI_w,
                          const float* __restrict__ WI_b,
                          _Float16* __restrict__ x16) {
  __shared__ float vin[IMG_DIM + U_DIM];
  const int t = threadIdx.x;
  const int u = *uid;
  for (int k = t; k < IMG_DIM + U_DIM; k += 256)
    vin[k] = (k < IMG_DIM) ? img[k] : wei_user[(size_t)u * U_DIM + (k - IMG_DIM)];
  __syncthreads();
  const int wave = t >> 5, lane = t & 31;
  const int row = blockIdx.x * 8 + wave;            // 64 blocks * 8 waves = 512 rows
  const float* wr = WI_w + (size_t)row * (IMG_DIM + U_DIM);
  float s = 0.0f;
  for (int k = lane; k < IMG_DIM + U_DIM; k += 32) s += wr[k] * vin[k];
#pragma unroll
  for (int off = 16; off > 0; off >>= 1) s += __shfl_xor(s, off, 32);
  if (lane == 0) x16[row] = (_Float16)(s + WI_b[row]);
}

__device__ __forceinline__ float sigmoidf_(float x) { return 1.0f / (1.0f + expf(-x)); }

// ------------------------------------------------------------------
// Core of the fused LSTM step: 4 gate mat-vecs via 4 independent WMMA
// chains + pointwise c/h update. Wave w owns h rows [w*16, w*16+16);
// gate-q tile index = q*64 + w. B fragments are loaded branchlessly:
// lanes whose B column != 0 only pollute D columns 1..15, never read.
// xlane = lane's x-fragment source (f16), already offset by hi*16.
// ------------------------------------------------------------------
template <bool STORE_H>
__device__ __forceinline__ void lstm_step_core(
    const _Float16* __restrict__ Aih, const _Float16* __restrict__ Ahh,
    const _Float16* __restrict__ xlane,      // per-lane x source (16 halves/chunk)
    const _Float16* __restrict__ hin,
    _Float16* __restrict__ hout16, float* __restrict__ cst,
    const float* __restrict__ bias, float* __restrict__ hout_f32,
    int w, int lane, int hi) {
  v8f acc0 = {}, acc1 = {}, acc2 = {}, acc3 = {};

  const _Float16* aX0 = Aih + ((size_t)(0 * 64 + w) * XCH * 32 + lane) * 16;
  const _Float16* aX1 = Aih + ((size_t)(1 * 64 + w) * XCH * 32 + lane) * 16;
  const _Float16* aX2 = Aih + ((size_t)(2 * 64 + w) * XCH * 32 + lane) * 16;
  const _Float16* aX3 = Aih + ((size_t)(3 * 64 + w) * XCH * 32 + lane) * 16;

  for (int c = 0; c < XCH; ++c) {
    v16h b = *(const v16h*)(xlane + c * 32);
    v16h a0 = *(const v16h*)aX0; aX0 += 512;
    v16h a1 = *(const v16h*)aX1; aX1 += 512;
    v16h a2 = *(const v16h*)aX2; aX2 += 512;
    v16h a3 = *(const v16h*)aX3; aX3 += 512;
    acc0 = __builtin_amdgcn_wmma_f32_16x16x32_f16(false, a0, false, b, (short)0, acc0, false, false);
    acc1 = __builtin_amdgcn_wmma_f32_16x16x32_f16(false, a1, false, b, (short)0, acc1, false, false);
    acc2 = __builtin_amdgcn_wmma_f32_16x16x32_f16(false, a2, false, b, (short)0, acc2, false, false);
    acc3 = __builtin_amdgcn_wmma_f32_16x16x32_f16(false, a3, false, b, (short)0, acc3, false, false);
  }

  const _Float16* aH0 = Ahh + ((size_t)(0 * 64 + w) * HCH * 32 + lane) * 16;
  const _Float16* aH1 = Ahh + ((size_t)(1 * 64 + w) * HCH * 32 + lane) * 16;
  const _Float16* aH2 = Ahh + ((size_t)(2 * 64 + w) * HCH * 32 + lane) * 16;
  const _Float16* aH3 = Ahh + ((size_t)(3 * 64 + w) * HCH * 32 + lane) * 16;

  for (int c = 0; c < HCH; ++c) {
    v16h b = *(const v16h*)(hin + c * 32 + hi * 16);  // branchless broadcast load
    v16h a0 = *(const v16h*)aH0; aH0 += 512;
    v16h a1 = *(const v16h*)aH1; aH1 += 512;
    v16h a2 = *(const v16h*)aH2; aH2 += 512;
    v16h a3 = *(const v16h*)aH3; aH3 += 512;
    acc0 = __builtin_amdgcn_wmma_f32_16x16x32_f16(false, a0, false, b, (short)0, acc0, false, false);
    acc1 = __builtin_amdgcn_wmma_f32_16x16x32_f16(false, a1, false, b, (short)0, acc1, false, false);
    acc2 = __builtin_amdgcn_wmma_f32_16x16x32_f16(false, a2, false, b, (short)0, acc2, false, false);
    acc3 = __builtin_amdgcn_wmma_f32_16x16x32_f16(false, a3, false, b, (short)0, acc3, false, false);
  }

  // column 0 of D: lane 0 holds M=0..7 (VGPR j), lane 16 holds M=8..15
  if (lane == 0 || lane == 16) {
    const int rbase = w * 16 + hi * 8;
#pragma unroll
    for (int j = 0; j < 8; ++j) {
      const int r = rbase + j;
      const float ig = acc0[j] + bias[r];
      const float fg = acc1[j] + bias[H_DIM + r];
      const float gg = acc2[j] + bias[2 * H_DIM + r];
      const float og = acc3[j] + bias[3 * H_DIM + r];
      const float cn = sigmoidf_(fg) * cst[r] + sigmoidf_(ig) * tanhf(gg);
      const float hn = sigmoidf_(og) * tanhf(cn);
      cst[r] = cn;
      hout16[r] = (_Float16)hn;
      if (STORE_H) hout_f32[r] = hn;
    }
  }
}

// Prime step: x comes from precomputed f16 x0; h output not recorded.
// grid: 64 blocks x 32 threads (1 wave).
__global__ void step_prime_kernel(const _Float16* __restrict__ Aih,
                                  const _Float16* __restrict__ Ahh,
                                  const _Float16* __restrict__ x16,
                                  const _Float16* __restrict__ hin,
                                  _Float16* __restrict__ hout16,
                                  float* __restrict__ cst,
                                  const float* __restrict__ bias) {
  const int w = blockIdx.x, lane = threadIdx.x, hi = lane >> 4;
  lstm_step_core<false>(Aih, Ahh, x16 + hi * 16, hin, hout16, cst, bias,
                        nullptr, w, lane, hi);
}

// Decode step: embed word_emb[wid] inline (f32 -> f16 into LDS, then the
// WMMA loop streams fragments from LDS).  grid: 64 blocks x 32 threads.
__global__ void step_decode_kernel(const _Float16* __restrict__ Aih,
                                   const _Float16* __restrict__ Ahh,
                                   const float* __restrict__ word_emb,
                                   const int* __restrict__ wid,
                                   const _Float16* __restrict__ hin,
                                   _Float16* __restrict__ hout16,
                                   float* __restrict__ cst,
                                   const float* __restrict__ bias,
                                   float* __restrict__ hout_f32) {
  const int w = blockIdx.x, lane = threadIdx.x, hi = lane >> 4;
  __shared__ _Float16 xs[W_DIM];
  const float* wrow = word_emb + (size_t)(*wid) * W_DIM;
  // 32 lanes cooperatively convert the 512-entry row (16 values per lane pass)
#pragma unroll
  for (int c = 0; c < XCH; c += 2) {    // 32 threads * 16 = 512 per pass
    const int base = (c + (lane >> 4)) * 32 + (lane & 15) * 2;
    xs[base]     = (_Float16)wrow[base];
    xs[base + 1] = (_Float16)wrow[base + 1];
  }
  __syncthreads();                      // single-wave WG: waits DScnt, barrier=NOP
  lstm_step_core<true>(Aih, Ahh, xs + hi * 16, hin, hout16, cst, bias,
                       hout_f32, w, lane, hi);
}

// ------------------------------------------------------------------
// logits = WP_pack @ h16 + WP_b ; per-wave argmax partial (first-index
// tie-break), then the last-arriving block reduces the 2000 partials
// (deterministic regardless of which block performs it) and emits the
// next token.  grid: 250 blocks x 256 threads.
// ------------------------------------------------------------------
__global__ void logits_argmax_kernel(const _Float16* __restrict__ P,
                                     const _Float16* __restrict__ h16,
                                     const float* __restrict__ WPb,
                                     float* __restrict__ pval,
                                     int* __restrict__ pidx,
                                     int* __restrict__ ticket,
                                     int* __restrict__ wid,
                                     float* __restrict__ cap_out) {
  const int tile = (blockIdx.x * blockDim.x + threadIdx.x) >> 5;  // 0..1999
  const int lane = threadIdx.x & 31;
  const int hi   = lane >> 4;
  v8f acc = {};
  const _Float16* ap = P + ((size_t)tile * HCH * 32 + lane) * 16;
  for (int c = 0; c < HCH; ++c) {
    v16h a = *(const v16h*)ap; ap += 512;
    v16h b = *(const v16h*)(h16 + c * 32 + hi * 16);  // branchless
    acc = __builtin_amdgcn_wmma_f32_16x16x32_f16(false, a, false, b, (short)0, acc, false, false);
  }
  float bestv = -INFINITY;
  int   besti = 0;
  if (lane == 0 || lane == 16) {
    const int base = tile * 16 + hi * 8;
#pragma unroll
    for (int j = 0; j < 8; ++j) {
      const float v = acc[j] + WPb[base + j];
      if (v > bestv) { bestv = v; besti = base + j; }    // strict > keeps first index
    }
  }
  const float ov = __shfl(bestv, 16, 32);
  const int   oi = __shfl(besti, 16, 32);
  if (lane == 0) {
    if (ov > bestv) { bestv = ov; besti = oi; }          // lane0 rows < lane16 rows
    pval[tile] = bestv;
    pidx[tile] = besti;
  }

  // ---- last block reduces all partials ----
  __shared__ int isLast;
  __syncthreads();                      // all partial stores issued
  if (threadIdx.x == 0) {
    __threadfence();                    // make them device-visible
    isLast = (atomicAdd(ticket, 1) == (int)gridDim.x - 1);
  }
  __syncthreads();
  if (isLast) {
    __shared__ float sv[256];
    __shared__ int   si[256];
    const int t = threadIdx.x;
    float bv = -INFINITY; int bi = 0x7fffffff;
    for (int i = t; i < NTILES_P; i += 256) {
      const float v = pval[i]; const int id = pidx[i];
      if (v > bv || (v == bv && id < bi)) { bv = v; bi = id; }
    }
    sv[t] = bv; si[t] = bi;
    __syncthreads();
    for (int s = 128; s > 0; s >>= 1) {
      if (t < s) {
        if (sv[t + s] > sv[t] || (sv[t + s] == sv[t] && si[t + s] < si[t])) {
          sv[t] = sv[t + s]; si[t] = si[t + s];
        }
      }
      __syncthreads();
    }
    if (t == 0) { *wid = si[0]; *cap_out = (float)si[0]; *ticket = 0; }
  }
}

// ------------------------------------------------------------------
extern "C" void kernel_launch(void* const* d_in, const int* in_sizes, int n_in,
                              void* d_out, int out_size, void* d_ws, size_t ws_size,
                              hipStream_t stream) {
  (void)in_sizes; (void)n_in; (void)out_size; (void)ws_size;

  const float* img      = (const float*)d_in[0];
  const int*   uid      = (const int*)  d_in[1];
  const float* wei_user = (const float*)d_in[2];
  const float* WI_w     = (const float*)d_in[3];
  const float* WI_b     = (const float*)d_in[4];
  const float* WP_w     = (const float*)d_in[5];
  const float* WP_b     = (const float*)d_in[6];
  const float* W_ih     = (const float*)d_in[7];
  const float* W_hh     = (const float*)d_in[8];
  const float* b_ih     = (const float*)d_in[9];
  const float* b_hh     = (const float*)d_in[10];
  const float* word_emb = (const float*)d_in[11];

  float* out = (float*)d_out;                // [64*1024 hiddens][64 captions]

  // ---- workspace carve-up (256B aligned) ----
  char* ws = (char*)d_ws;
  size_t off = 0;
  auto carve = [&](size_t bytes) {
    char* p = ws + off;
    off += (bytes + 255) & ~(size_t)255;
    return p;
  };
  _Float16* wihP = (_Float16*)carve((size_t)256  * 16 * 512 * 2); //  4 MiB
  _Float16* whhP = (_Float16*)carve((size_t)256  * 32 * 512 * 2); //  8 MiB
  _Float16* wpP  = (_Float16*)carve((size_t)2000 * 32 * 512 * 2); // ~62.5 MiB
  _Float16* x16  = (_Float16*)carve(W_DIM * 2);
  _Float16* hA   = (_Float16*)carve(H_DIM * 2);
  _Float16* hB   = (_Float16*)carve(H_DIM * 2);
  float*    cst  = (float*)   carve(H_DIM * 4);
  float*    bias = (float*)   carve(GATES * 4);
  float*    pval = (float*)   carve(NTILES_P * 4);
  int*      pidx = (int*)     carve(NTILES_P * 4);
  int*      wid  = (int*)     carve(4);
  int*      tick = (int*)     carve(4);
  _Float16* hbuf[2] = {hA, hB};

  // ---- one-time (per call) f16 pack of the L2-resident weights ----
  pack_w_kernel<<<512,  256, 0, stream>>>(W_ih, wihP, GATES, W_DIM);   // 4096 waves
  pack_w_kernel<<<1024, 256, 0, stream>>>(W_hh, whhP, GATES, H_DIM);   // 8192 waves
  pack_w_kernel<<<8000, 256, 0, stream>>>(WP_w, wpP,  VOC,   H_DIM);   // 64000 waves

  prep_kernel<<<16, 256, 0, stream>>>(b_ih, b_hh, bias, hA, cst, wid, tick);
  x0_kernel<<<64, 256, 0, stream>>>(img, wei_user, uid, WI_w, WI_b, x16);

  // priming LSTM step: x0 path, reads hA (zeros), writes hB, h not recorded
  step_prime_kernel<<<64, 32, 0, stream>>>(wihP, whhP, x16, hbuf[0], hbuf[1],
                                           cst, bias);

  // 64 sequential decode steps: 2 launches per step
  for (int t = 0; t < CAP_LEN; ++t) {
    _Float16* hin  = hbuf[(t + 1) & 1];
    _Float16* hout = hbuf[t & 1];
    step_decode_kernel<<<64, 32, 0, stream>>>(wihP, whhP, word_emb, wid,
                                              hin, hout, cst, bias,
                                              out + (size_t)t * H_DIM);
    logits_argmax_kernel<<<250, 256, 0, stream>>>(wpP, hout, WP_b, pval, pidx,
                                                  tick, wid,
                                                  out + (size_t)CAP_LEN * H_DIM + t);
  }
}
